// SharedGraphLayer_120259085137
// MI455X (gfx1250) — matrix operand order repaired and checked by reading, not compiled, not run on hardware
//
#include <hip/hip_runtime.h>

// Sizes from the reference
#define Bn   8
#define Gn   8
#define Vn   1024
#define FIN  64
#define FOUT 64
#define GF   512      // G * FOUT
#define ALPHA 0.2f

typedef __attribute__((ext_vector_type(16))) _Float16 v16h;
typedef __attribute__((ext_vector_type(8)))  _Float16 h8;
typedef __attribute__((ext_vector_type(4)))  _Float16 h4;
typedef __attribute__((ext_vector_type(8)))  float    v8f;

__device__ __forceinline__ float leaky(float x) { return x >= 0.f ? x : ALPHA * x; }

// CDNA5 async memory->LDS copy (VGLOBAL GLOBAL_LOAD_ASYNC_TO_LDS_B128, ASYNCcnt)
// vdst operand = per-lane LDS byte offset; vaddr = per-lane 64-bit global address.
__device__ __forceinline__ void async_ld_b128(unsigned int lds_off, unsigned long long gaddr) {
    asm volatile("global_load_async_to_lds_b128 %0, %1, off"
                 :: "v"(lds_off), "v"(gaddr) : "memory");
}
#define WAIT_ASYNC(n) asm volatile("s_wait_asynccnt " #n ::: "memory")

// ---------------------------------------------------------------------------
// K1: a[b,j,v] = (sum_g adj[b,g,j,v]) >= 1, stored as f16 (exact 0/1)
// Vectorized: 4 consecutive v per thread via b128 loads (256 MiB stream ->
// this kernel dominates total runtime; minimize instructions per byte).
// ---------------------------------------------------------------------------
__global__ void k_adj_reduce(const int* __restrict__ adj, _Float16* __restrict__ a16) {
    int idx4 = blockIdx.x * blockDim.x + threadIdx.x;    // B*V*V/4 = 2Mi
    int idx  = idx4 << 2;
    int b  = idx >> 20;
    int jv = idx & ((1 << 20) - 1);
    const int4* p = (const int4*)(adj + (size_t)b * Gn * Vn * Vn + jv);
    int4 s = {0, 0, 0, 0};
#pragma unroll
    for (int g = 0; g < Gn; ++g) {
        int4 t = p[(size_t)g * (Vn * Vn / 4)];
        s.x += t.x; s.y += t.y; s.z += t.z; s.w += t.w;
    }
    h4 r;
    r[0] = (_Float16)(s.x >= 1 ? 1.0f : 0.0f);
    r[1] = (_Float16)(s.y >= 1 ? 1.0f : 0.0f);
    r[2] = (_Float16)(s.z >= 1 ? 1.0f : 0.0f);
    r[3] = (_Float16)(s.w >= 1 ? 1.0f : 0.0f);
    *(h4*)(a16 + idx) = r;
}

// ---------------------------------------------------------------------------
// K2: deg[b,v] = mean_j a[b,j,v]   (column means)
// ---------------------------------------------------------------------------
__global__ void k_deg(const _Float16* __restrict__ a16, float* __restrict__ deg) {
    int i = blockIdx.x * blockDim.x + threadIdx.x;       // B*V = 8192
    int b = i >> 10, v = i & 1023;
    const _Float16* p = a16 + ((size_t)b << 20) + v;
    float s = 0.f;
    for (int j = 0; j < Vn; ++j) s += (float)p[j << 10];
    deg[i] = s * (1.0f / Vn);
}

// ---------------------------------------------------------------------------
// K3: Wh[b,v,g*64+o] = leaky( sum_i h[b,g,v,i] * W[g,i,o] )  via WMMA f16
// grid: (16 mblocks, G, B), 128 threads (4 waves); wave -> one 16-row M tile
// ---------------------------------------------------------------------------
__global__ __launch_bounds__(128) void k_wh(const float* __restrict__ h,
                                            const float* __restrict__ W,
                                            float* __restrict__ wh32,
                                            _Float16* __restrict__ wh16) {
    int wid  = threadIdx.x >> 5;
    int lane = threadIdx.x & 31;
    int g = blockIdx.y, b = blockIdx.z;
    int m0 = blockIdx.x * 64 + wid * 16;
    int lm = lane & 15;
    int hi = (lane >> 4) & 1;

    v8f acc[4] = {};
    const float* hrow = h + ((size_t)(b * Gn + g) * Vn + m0 + lm) * FIN;

#pragma unroll
    for (int k0 = 0; k0 < FIN; k0 += 32) {
        v16h a;
        int kb = hi ? 16 : 0;
#pragma unroll
        for (int i = 0; i < 8; ++i) {
            a[i]     = (_Float16)hrow[k0 + kb + i];
            a[8 + i] = (_Float16)hrow[k0 + 8 + kb + i];
        }
        int kB = hi ? 16 : 0;
#pragma unroll
        for (int nt = 0; nt < 4; ++nt) {
            int o = nt * 16 + lm;
            v16h bf;
#pragma unroll
            for (int i = 0; i < 16; ++i)
                bf[i] = (_Float16)W[((size_t)g * FIN + k0 + kB + i) * FOUT + o];
            acc[nt] = __builtin_amdgcn_wmma_f32_16x16x32_f16(
                false, a, false, bf, (short)0, acc[nt], false, false);
        }
    }
#pragma unroll
    for (int nt = 0; nt < 4; ++nt) {
#pragma unroll
        for (int r = 0; r < 8; ++r) {
            float val = leaky(acc[nt][r]);
            int v = m0 + r + hi * 8;
            int c = g * FOUT + nt * 16 + lm;
            size_t idx = (size_t)(b * Vn + v) * GF + c;
            wh32[idx] = val;
            wh16[idx] = (_Float16)val;
        }
    }
}

// ---------------------------------------------------------------------------
// K4: feature_out = leaky( (a @ Wh) / V ), written directly in [B,G,V,FOUT]
// Double-buffered LDS WMMA GEMM: block tile 128x64, 8 waves of 32x32, K=32.
//  - A tile (128x32 halves) staged via global_load_async_to_lds_b128 (ASYNCcnt)
//  - B tile (32x64) staged transposed so fragment reads are contiguous b128
// ---------------------------------------------------------------------------
#define ASTR  48   // A LDS row stride (halves): 96B rows, 16B-aligned chunks
#define BSTRT 40   // transposed-B LDS row stride (halves): 80B rows (5*16B)
__global__ __launch_bounds__(256) void k_gemm_aw(const _Float16* __restrict__ a16,
                                                 const _Float16* __restrict__ wh16,
                                                 float* __restrict__ fo) {
    __shared__ __align__(16) _Float16 As[2][128 * ASTR];
    __shared__ __align__(16) _Float16 Bt[2][64 * BSTRT];

    int t = threadIdx.x;
    int wid = t >> 5, lane = t & 31;
    int b  = blockIdx.z;
    int mb = blockIdx.y * 128;
    int nb = blockIdx.x * 64;
    int wm = (wid & 3) * 32;
    int wn = (wid >> 2) * 32;
    int lm = lane & 15;
    int hi = (lane >> 4) & 1;

    v8f acc[2][2] = {};

    // staging maps: A 128x32 halves (2 threads/row), B 32x64 halves (8 threads/row)
    int arow = t >> 1, acol = (t & 1) * 16;
    int brow = t >> 3, bcol = (t & 7) * 8;
    const _Float16* gA = a16 + ((size_t)(b * Vn + mb + arow)) * Vn + acol;
    const _Float16* gB = wh16 + ((size_t)(b * Vn + brow)) * GF + nb + bcol;

    // ---- prologue: stage tile 0 into buffer 0
    {
        unsigned int d0 = (unsigned int)(size_t)&As[0][arow * ASTR + acol];
        async_ld_b128(d0,      (unsigned long long)(size_t)gA);
        async_ld_b128(d0 + 16, (unsigned long long)(size_t)(gA + 8));
        h8 d = *(const h8*)gB;
#pragma unroll
        for (int i = 0; i < 8; ++i) Bt[0][(bcol + i) * BSTRT + brow] = d[i];
    }

    for (int k0 = 0; k0 < Vn; k0 += 32) {
        int cur = (k0 >> 5) & 1;
        int nxt = cur ^ 1;
        if (k0 + 32 < Vn) {   // uniform branch: prefetch next tile
            unsigned int d0 = (unsigned int)(size_t)&As[nxt][arow * ASTR + acol];
            async_ld_b128(d0,      (unsigned long long)(size_t)(gA + k0 + 32));
            async_ld_b128(d0 + 16, (unsigned long long)(size_t)(gA + k0 + 40));
            h8 d = *(const h8*)(gB + (size_t)(k0 + 32) * GF);
#pragma unroll
            for (int i = 0; i < 8; ++i) Bt[nxt][(bcol + i) * BSTRT + brow] = d[i];
            WAIT_ASYNC(2);    // retire the 2 async loads of tile k
        } else {
            WAIT_ASYNC(0);
        }
        __syncthreads();

        v16h afr[2], bfr[2];
#pragma unroll
        for (int mt = 0; mt < 2; ++mt) {
            const _Float16* Ar = &As[cur][(wm + mt * 16 + lm) * ASTR + hi * 16];
            h8 lo = *(const h8*)Ar;
            h8 hh = *(const h8*)(Ar + 8);
#pragma unroll
            for (int i = 0; i < 8; ++i) { afr[mt][i] = lo[i]; afr[mt][8 + i] = hh[i]; }
        }
#pragma unroll
        for (int nt = 0; nt < 2; ++nt) {
            const _Float16* Br = &Bt[cur][(wn + nt * 16 + lm) * BSTRT + hi * 16];
            h8 lo = *(const h8*)Br;
            h8 hh = *(const h8*)(Br + 8);
#pragma unroll
            for (int i = 0; i < 8; ++i) { bfr[nt][i] = lo[i]; bfr[nt][8 + i] = hh[i]; }
        }
#pragma unroll
        for (int mt = 0; mt < 2; ++mt)
#pragma unroll
            for (int nt = 0; nt < 2; ++nt)
                acc[mt][nt] = __builtin_amdgcn_wmma_f32_16x16x32_f16(
                    false, afr[mt], false, bfr[nt], (short)0, acc[mt][nt], false, false);
        __syncthreads();
    }

    const float scale = 1.0f / Vn;
#pragma unroll
    for (int mt = 0; mt < 2; ++mt)
#pragma unroll
        for (int nt = 0; nt < 2; ++nt)
#pragma unroll
            for (int r = 0; r < 8; ++r) {
                float val = leaky(acc[mt][nt][r] * scale);
                int v = mb + wm + mt * 16 + r + hi * 8;
                int c = nb + wn + nt * 16 + lm;
                int g = c >> 6, o = c & 63;
                fo[(((size_t)(b * Gn + g)) * Vn + v) * FOUT + o] = val;
            }
}

// ---------------------------------------------------------------------------
// K5: feature_in = leaky(deg[b,v] * Wh[b,v,c]), written in [B,G,V,FOUT]
// ---------------------------------------------------------------------------
__global__ void k_featin(const float* __restrict__ wh32, const float* __restrict__ deg,
                         float* __restrict__ fin) {
    int idx = blockIdx.x * blockDim.x + threadIdx.x;     // B*V*GF = 4Mi
    int b   = idx >> 19;                                  // V*GF = 2^19
    int rem = idx & ((1 << 19) - 1);
    int v = rem >> 9;
    int c = rem & 511;
    float val = leaky(deg[(b << 10) + v] * wh32[idx]);
    int g = c >> 6, o = c & 63;
    fin[(((size_t)(b * Gn + g)) * Vn + v) * FOUT + o] = val;
}

// ---------------------------------------------------------------------------
// K6: both 5x5 convs (pad H by 2, none on W) + concat. blockIdx.z selects conv.
// grid: (B*G, V, 2), 64 threads (x = output column, active x<60)
// ---------------------------------------------------------------------------
__global__ __launch_bounds__(64) void k_conv(const float* __restrict__ fin,
                                             const float* __restrict__ fo,
                                             const float* __restrict__ w1,
                                             const float* __restrict__ b1,
                                             const float* __restrict__ w2,
                                             const float* __restrict__ b2,
                                             float* __restrict__ out) {
    __shared__ float ws[Gn * 25];
    int sel = blockIdx.z;
    int bg  = blockIdx.x;
    int b = bg >> 3, go = bg & 7;
    int y = blockIdx.y;
    int x = threadIdx.x;

    const float* w    = (sel ? w2 : w1) + go * Gn * 25;
    const float* bias = sel ? b2 : b1;
    const float* X    = sel ? fo : fin;

    for (int i = x; i < Gn * 25; i += 64) ws[i] = w[i];
    __syncthreads();
    if (x >= 60) return;

    float acc = bias[go];
    for (int gi = 0; gi < Gn; ++gi) {
        const float* Xg = X + ((size_t)(b * Gn + gi)) * Vn * FOUT;
        const float* wg = &ws[gi * 25];
#pragma unroll
        for (int ky = 0; ky < 5; ++ky) {
            int yy = y + ky - 2;
            if (yy < 0 || yy >= Vn) continue;
            const float* row = Xg + (size_t)yy * FOUT + x;
#pragma unroll
            for (int kx = 0; kx < 5; ++kx) acc += row[kx] * wg[ky * 5 + kx];
        }
    }
    out[(((size_t)(b * Gn + go)) * Vn + y) * 120 + sel * 60 + x] = acc;
}

// ---------------------------------------------------------------------------
extern "C" void kernel_launch(void* const* d_in, const int* in_sizes, int n_in,
                              void* d_out, int out_size, void* d_ws, size_t ws_size,
                              hipStream_t stream) {
    const float* h   = (const float*)d_in[0];
    const int*   adj = (const int*)d_in[1];
    const float* W   = (const float*)d_in[2];
    const float* c1w = (const float*)d_in[3];
    const float* c1b = (const float*)d_in[4];
    const float* c2w = (const float*)d_in[5];
    const float* c2b = (const float*)d_in[6];
    float* out = (float*)d_out;

    char* ws = (char*)d_ws;
    _Float16* a16  = (_Float16*)(ws);                       // 16 MiB
    _Float16* wh16 = (_Float16*)(ws + (16u << 20));         //  8 MiB
    float*    wh32 = (float*)(ws + (24u << 20));            // 16 MiB
    float*    deg  = (float*)(ws + (40u << 20));            // 32 KiB
    float*    fin  = (float*)(ws + (41u << 20));            // 16 MiB
    float*    fo   = (float*)(ws + (57u << 20));            // 16 MiB

    k_adj_reduce<<<dim3((Bn * Vn * Vn / 4) / 256), 256, 0, stream>>>(adj, a16);
    k_deg<<<dim3((Bn * Vn) / 256), 256, 0, stream>>>(a16, deg);
    k_wh<<<dim3(Vn / 64, Gn, Bn), 128, 0, stream>>>(h, W, wh32, wh16);
    k_featin<<<dim3((Bn * Vn * GF) / 256), 256, 0, stream>>>(wh32, deg, fin);
    k_gemm_aw<<<dim3(GF / 64, Vn / 128, Bn), 256, 0, stream>>>(a16, wh16, fo);
    k_conv<<<dim3(Bn * Gn, Vn, 2), 64, 0, stream>>>(fin, fo, c1w, c1b, c2w, c2b, out);
}
